// SelfAttention_24532853195492
// MI455X (gfx1250) — matrix operand order repaired and checked by reading, not compiled, not run on hardware
//
#include <hip/hip_runtime.h>
#include <hip/hip_bf16.h>
#include <math.h>

// Problem constants (from reference): B=2, T=2048, C=1024, H=16, HD=64
#define DB 2
#define DT 2048
#define DC 1024
#define DH 16
#define DHD 64
#define DM (DB*DT)      // 4096 rows
#define DN3 (3*DC)      // 3072

typedef __attribute__((ext_vector_type(16))) _Float16 v16h;
typedef __attribute__((ext_vector_type(8)))  _Float16 v8h;
typedef __attribute__((ext_vector_type(8)))  float    v8f;

#define WMMA_F32_F16(a,b,c) \
  __builtin_amdgcn_wmma_f32_16x16x32_f16(false,(a),false,(b),(short)0,(c),false,false)

// ---- fragment loaders: every fragment = two 128-bit LDS loads -------------------
// A (16x32 f16): lane -> row M=lane%16, half=lane/16,
// elements j = K[half*16 + j] (same k order as the B fragments).
static __device__ __forceinline__ v16h frag_a(const _Float16* row, int half) {
  v8h lo = *(const v8h*)(row + half * 16);
  v8h hi = *(const v8h*)(row + half * 16 + 8);
  return __builtin_shufflevector(lo, hi, 0,1,2,3,4,5,6,7,8,9,10,11,12,13,14,15);
}
// B (32x16 f16): lane -> col N=lane%16, half=lane/16, elements j = K[half*16 + j].
// With the tile staged N-major, the 16 halves are contiguous at &Bt[n*ld + half*16].
static __device__ __forceinline__ v16h frag_contig(const _Float16* p) {
  v8h lo = *(const v8h*)(p);
  v8h hi = *(const v8h*)(p + 8);
  return __builtin_shufflevector(lo, hi, 0,1,2,3,4,5,6,7,8,9,10,11,12,13,14,15);
}

// ---- DPP16 row reductions (16-lane groups == C/D fragment row striping) ---------
// row_ror:k rotates within each 16-lane row; butterfly windows 2,4,8,16 cover all.
template <int CTRL>
static __device__ __forceinline__ float dpp_rot(float x) {
  int xi = __float_as_int(x);
  int r  = __builtin_amdgcn_update_dpp(xi, xi, CTRL, 0xf, 0xf, false);
  return __int_as_float(r);
}
static __device__ __forceinline__ float row16_max(float x) {
  x = fmaxf(x, dpp_rot<0x121>(x));   // row_ror:1
  x = fmaxf(x, dpp_rot<0x122>(x));   // row_ror:2
  x = fmaxf(x, dpp_rot<0x124>(x));   // row_ror:4
  x = fmaxf(x, dpp_rot<0x128>(x));   // row_ror:8
  return x;
}
static __device__ __forceinline__ float row16_sum(float x) {
  x += dpp_rot<0x121>(x);
  x += dpp_rot<0x122>(x);
  x += dpp_rot<0x124>(x);
  x += dpp_rot<0x128>(x);
  return x;
}

// ---------------- f32 -> f16 cast ------------------------------------------------
__global__ void cast_f32_to_f16(const float* __restrict__ in, _Float16* __restrict__ out,
                                size_t n) {
  size_t i = (size_t)blockIdx.x * blockDim.x + threadIdx.x;
  if (i < n) out[i] = (_Float16)in[i];
}

// ---------------- tiled WMMA GEMM: Out[M,N] = A[M,K] * Bh[K,N] + bias[N] ---------
// Block tile 128x128, BK=32. 256 threads = 8 waves in 4x2 grid; wave tile 32x64.
template <bool A_IS_F16>
__global__ __launch_bounds__(256)
void gemm_wmma(const void* __restrict__ Ap, const _Float16* __restrict__ Bh,
               const float* __restrict__ bias, float* __restrict__ Out,
               int M, int N, int K) {
  constexpr int LDA = 40;   // 32 + 8 pad halves (80 B rows, 16B aligned)
  constexpr int LDB = 40;   // Bt is N-major: 128 rows(n) x 32(k) + 8 pad
  __shared__ __align__(16) _Float16 As[128 * LDA];
  __shared__ __align__(16) _Float16 Bt[128 * LDB];

  const int tid  = threadIdx.x;
  const int lane = tid & 31;
  const int wid  = tid >> 5;
  const int wr   = wid >> 1;   // 0..3
  const int wc   = wid & 1;    // 0..1
  const int ml   = lane & 15;
  const int half = lane >> 4;
  const int brow = blockIdx.y * 128;
  const int bcol = blockIdx.x * 128;

  const float*    Af = (const float*)Ap;
  const _Float16* Ah = (const _Float16*)Ap;

  // per-thread staging coordinates
  const int ar  = tid >> 1;             // A: row 0..127
  const int ac0 = (tid & 1) * 16;       // A: col chunk 0/16
  const int bk  = tid >> 3;             // B: k row 0..31
  const int bn0 = (tid & 7) * 16;       // B: n chunk 0..112

  v8f acc[2][4] = {};

  for (int k0 = 0; k0 < K; k0 += 32) {
    // ---- stage A tile 128x32 (row-major, convert f32->f16 if needed) ----
    {
      size_t g0 = (size_t)(brow + ar) * K + (k0 + ac0);
      if (A_IS_F16) {
        *(v8h*)&As[ar * LDA + ac0]     = *(const v8h*)&Ah[g0];
        *(v8h*)&As[ar * LDA + ac0 + 8] = *(const v8h*)&Ah[g0 + 8];
      } else {
        _Float16 tmpa[16];
#pragma unroll
        for (int i = 0; i < 16; ++i) tmpa[i] = (_Float16)Af[g0 + i];
        *(v8h*)&As[ar * LDA + ac0]     = *(const v8h*)&tmpa[0];
        *(v8h*)&As[ar * LDA + ac0 + 8] = *(const v8h*)&tmpa[8];
      }
      // prefetch next A tile (global_prefetch_b8)
      if (k0 + 32 < K)
        __builtin_prefetch(A_IS_F16 ? (const void*)&Ah[g0 + 32]
                                    : (const void*)&Af[g0 + 32], 0, 3);
    }
    // ---- stage B tile 32x128 transposed into Bt[n][k] ----
    {
      size_t gb = (size_t)(k0 + bk) * N + (bcol + bn0);
      _Float16 tmpb[16];
      *(v8h*)&tmpb[0] = *(const v8h*)&Bh[gb];
      *(v8h*)&tmpb[8] = *(const v8h*)&Bh[gb + 8];
#pragma unroll
      for (int j = 0; j < 16; ++j) Bt[(bn0 + j) * LDB + bk] = tmpb[j];
      if (k0 + 32 < K)
        __builtin_prefetch(&Bh[gb + (size_t)32 * N], 0, 3);
    }
    __syncthreads();

    v16h afr[2], bfr[4];
#pragma unroll
    for (int rs = 0; rs < 2; ++rs)
      afr[rs] = frag_a(&As[(wr * 32 + rs * 16 + ml) * LDA], half);
#pragma unroll
    for (int cs = 0; cs < 4; ++cs)
      bfr[cs] = frag_contig(&Bt[(wc * 64 + cs * 16 + ml) * LDB + half * 16]);

#pragma unroll
    for (int rs = 0; rs < 2; ++rs)
#pragma unroll
      for (int cs = 0; cs < 4; ++cs)
        acc[rs][cs] = WMMA_F32_F16(afr[rs], bfr[cs], acc[rs][cs]);

    __syncthreads();
  }

  // epilogue: C/D layout -> lane holds (M = r + 8*half, N = ml) for vgpr r
#pragma unroll
  for (int rs = 0; rs < 2; ++rs) {
#pragma unroll
    for (int cs = 0; cs < 4; ++cs) {
      int col = bcol + wc * 64 + cs * 16 + ml;
      float bv = bias[col];
#pragma unroll
      for (int r = 0; r < 8; ++r) {
        int row = brow + wr * 32 + rs * 16 + half * 8 + r;
        Out[(size_t)row * N + col] = acc[rs][cs][r] + bv;
      }
    }
  }
}

// ---------------- RoPE + head split: qkv f32 [B,T,3C] -> q,k,v f16 [B,H,T,HD] ----
__global__ __launch_bounds__(256)
void rope_split(const float* __restrict__ qkv, _Float16* __restrict__ qh,
                _Float16* __restrict__ kh, _Float16* __restrict__ vh) {
  size_t idx = (size_t)blockIdx.x * blockDim.x + threadIdx.x; // 2^21 threads
  int pair = idx & 31;              // HD/2
  int h    = (idx >> 5) & 15;
  int t    = (idx >> 9) & 2047;
  int b    = (int)(idx >> 20);

  const float* row = qkv + (size_t)(b * DT + t) * DN3;
  const int coff = h * DHD;
  const int d0 = 2 * pair, d1 = d0 + 1;

  float q0 = row[coff + d0],            q1 = row[coff + d1];
  float k0 = row[DC + coff + d0],       k1 = row[DC + coff + d1];
  float v0 = row[2 * DC + coff + d0],   v1 = row[2 * DC + coff + d1];

  // inv_freq = 10000^(-2*pair/HD)
  float invf = __expf(-9.210340371976184f * (float)(2 * pair) / (float)DHD);
  float f = (float)t * invf;
  float cs = cosf(f), sn = sinf(f);

  size_t o = (((size_t)(b * DH + h)) * DT + t) * DHD;
  qh[o + d0] = (_Float16)(q0 * cs - q1 * sn);
  qh[o + d1] = (_Float16)(q1 * cs + q0 * sn);
  kh[o + d0] = (_Float16)(k0 * cs - k1 * sn);
  kh[o + d1] = (_Float16)(k1 * cs + k0 * sn);
  vh[o + d0] = (_Float16)v0;
  vh[o + d1] = (_Float16)v1;
}

// ---------------- flash attention: per (b,h, 64-row q tile) ----------------------
// 128 threads = 4 waves; wave handles 16 q rows. 32-key blocks, online softmax.
// K tile staged in natural [key][hd] layout (== transposed staging of K^T, so the
// QK^T B-fragment is two contiguous b128 LDS loads); V staged transposed [hd][key].
__global__ __launch_bounds__(128)
void flash_attn(const _Float16* __restrict__ qh, const _Float16* __restrict__ kh,
                const _Float16* __restrict__ vh, _Float16* __restrict__ y) {
  const int bh = blockIdx.y;           // b*H + h
  const int qt = blockIdx.x;           // q tile of 64 rows
  const int b = bh >> 4, h = bh & 15;
  const int tid = threadIdx.x, lane = tid & 31, wid = tid >> 5;
  const int ml = lane & 15, half = lane >> 4;

  const _Float16* Qp = qh + (size_t)bh * DT * DHD;
  const _Float16* Kp = kh + (size_t)bh * DT * DHD;
  const _Float16* Vp = vh + (size_t)bh * DT * DHD;

  __shared__ __align__(16) _Float16 Qs[64 * 72];   // [qrow][hd]   (+8 pad)
  __shared__ __align__(16) _Float16 Ks[32 * 72];   // [key][hd]    (+8 pad)
  __shared__ __align__(16) _Float16 Vt[64 * 40];   // [hd][key]    (+8 pad)
  __shared__ __align__(16) _Float16 Ps[4 * 16 * 40]; // per-wave 16x32 P

  // ---- stage Q tile 64x64 (32 contiguous halves per thread) ----
  {
    int r = tid >> 1, c0 = (tid & 1) * 32;
    const _Float16* src = &Qp[(size_t)(qt * 64 + r) * DHD + c0];
    _Float16* dst = &Qs[r * 72 + c0];
#pragma unroll
    for (int j = 0; j < 4; ++j)
      *(v8h*)(dst + 8 * j) = *(const v8h*)(src + 8 * j);
  }
  __syncthreads();

  v16h qa[2];
#pragma unroll
  for (int kk = 0; kk < 2; ++kk)
    qa[kk] = frag_a(&Qs[(wid * 16 + ml) * 72 + kk * 32], half);

  v8f o[4] = {};
  float mrow[8], lrow[8];
#pragma unroll
  for (int r = 0; r < 8; ++r) { mrow[r] = -1.0e30f; lrow[r] = 0.f; }

  const int qmaxBlock = qt * 64 + 63;
  const float scale = 0.125f;          // 1/sqrt(64)
  const int qbase = qt * 64 + wid * 16 + half * 8;

  // staging coords: 16 halves per thread for both K and V tiles
  const int skey = tid >> 2;           // 0..31
  const int sc0  = (tid & 3) * 16;     // 0..48

  for (int kb = 0; kb <= qmaxBlock; kb += 32) {
    // ---- stage K tile (natural) and V tile (transposed) ----
    {
      const _Float16* ksrc = &Kp[(size_t)(kb + skey) * DHD + sc0];
      *(v8h*)&Ks[skey * 72 + sc0]     = *(const v8h*)(ksrc);
      *(v8h*)&Ks[skey * 72 + sc0 + 8] = *(const v8h*)(ksrc + 8);

      _Float16 tmpv[16];
      const _Float16* vsrc = &Vp[(size_t)(kb + skey) * DHD + sc0];
      *(v8h*)&tmpv[0] = *(const v8h*)(vsrc);
      *(v8h*)&tmpv[8] = *(const v8h*)(vsrc + 8);
#pragma unroll
      for (int j = 0; j < 16; ++j) Vt[(sc0 + j) * 40 + skey] = tmpv[j];

      if (kb + 32 <= qmaxBlock) {      // prefetch next key block
        __builtin_prefetch(ksrc + 32 * DHD, 0, 3);
        __builtin_prefetch(vsrc + 32 * DHD, 0, 3);
      }
    }
    __syncthreads();

    // ---- S = Q * K^T : load all 4 B fragments, then 4 WMMAs ----
    v16h kb00 = frag_contig(&Ks[(ml)      * 72 +      half * 16]);
    v16h kb01 = frag_contig(&Ks[(ml)      * 72 + 32 + half * 16]);
    v16h kb10 = frag_contig(&Ks[(16 + ml) * 72 +      half * 16]);
    v16h kb11 = frag_contig(&Ks[(16 + ml) * 72 + 32 + half * 16]);
    v8f s0 = {}, s1 = {};
    s0 = WMMA_F32_F16(qa[0], kb00, s0);
    s0 = WMMA_F32_F16(qa[1], kb01, s0);
    s1 = WMMA_F32_F16(qa[0], kb10, s1);
    s1 = WMMA_F32_F16(qa[1], kb11, s1);

    // ---- online softmax (DPP row_ror butterfly over the 16-lane N groups) ----
    float p0[8], p1[8];
#pragma unroll
    for (int r = 0; r < 8; ++r) {
      int qr = qbase + r;
      float a0 = s0[r] * scale, a1 = s1[r] * scale;
      if (kb + ml      > qr) a0 = -1.0e30f;   // causal mask
      if (kb + 16 + ml > qr) a1 = -1.0e30f;
      float mx = row16_max(fmaxf(a0, a1));
      float mnew = fmaxf(mrow[r], mx);
      float sc = __expf(mrow[r] - mnew);
      p0[r] = __expf(a0 - mnew);
      p1[r] = __expf(a1 - mnew);
      float rs = row16_sum(p0[r] + p1[r]);
      lrow[r] = lrow[r] * sc + rs;
      mrow[r] = mnew;
#pragma unroll
      for (int fgt = 0; fgt < 4; ++fgt) o[fgt][r] *= sc;
    }

    // ---- stage P (C/D layout -> row-major LDS), reload as A fragment ----
    _Float16* pw = &Ps[wid * 640];
#pragma unroll
    for (int r = 0; r < 8; ++r) {
      pw[(half * 8 + r) * 40 + ml]      = (_Float16)p0[r];
      pw[(half * 8 + r) * 40 + 16 + ml] = (_Float16)p1[r];
    }
    asm volatile("s_wait_dscnt 0" ::: "memory");   // intra-wave LDS RAW (split counters)
    v16h pa = frag_a(&pw[ml * 40], half);

    // ---- O += P * V : load all 4 V fragments, then 4 WMMAs ----
    v16h vb[4];
#pragma unroll
    for (int fgt = 0; fgt < 4; ++fgt)
      vb[fgt] = frag_contig(&Vt[(fgt * 16 + ml) * 40 + half * 16]);
#pragma unroll
    for (int fgt = 0; fgt < 4; ++fgt)
      o[fgt] = WMMA_F32_F16(pa, vb[fgt], o[fgt]);
    __syncthreads();
  }

  // ---- finalize: y[b][t][h*64 + n]  (f16 for the projection GEMM) ----
#pragma unroll
  for (int r = 0; r < 8; ++r) {
    float inv = 1.0f / lrow[r];
    int t = qt * 64 + wid * 16 + half * 8 + r;
    size_t orow = ((size_t)(b * DT + t)) * DC + h * DHD;
#pragma unroll
    for (int fgt = 0; fgt < 4; ++fgt)
      y[orow + fgt * 16 + ml] = (_Float16)(o[fgt][r] * inv);
  }
}

// ---------------- launch ---------------------------------------------------------
extern "C" void kernel_launch(void* const* d_in, const int* in_sizes, int n_in,
                              void* d_out, int out_size, void* d_ws, size_t ws_size,
                              hipStream_t stream) {
  const float* x      = (const float*)d_in[0];
  const float* W_attn = (const float*)d_in[1];
  const float* b_attn = (const float*)d_in[2];
  const float* W_proj = (const float*)d_in[3];
  const float* b_proj = (const float*)d_in[4];
  // d_in[5] attn_mask: causal mask is applied implicitly
  float* out = (float*)d_out;
  char*  ws  = (char*)d_ws;

  const size_t szWah = (size_t)DC * DN3 * sizeof(_Float16);   //  6 MB
  const size_t szWph = (size_t)DC * DC  * sizeof(_Float16);   //  2 MB
  const size_t szQkv = (size_t)DM * DN3 * sizeof(float);      // 48 MB
  const size_t szHd  = (size_t)DM * DC  * sizeof(_Float16);   //  8 MB each

  size_t offWah = 0;
  size_t offWph = offWah + szWah;
  size_t offQkv = offWph + szWph;
  size_t offQh  = offQkv + szQkv;
  size_t offKh  = offQh  + szHd;
  size_t offVh  = offKh  + szHd;
  if (ws_size < offVh + szHd) return;

  _Float16* Wah = (_Float16*)(ws + offWah);
  _Float16* Wph = (_Float16*)(ws + offWph);
  float*    qkv = (float*)   (ws + offQkv);
  _Float16* yh  = (_Float16*)(ws + offQkv);   // reuse qkv region after RoPE
  _Float16* qhp = (_Float16*)(ws + offQh);
  _Float16* khp = (_Float16*)(ws + offKh);
  _Float16* vhp = (_Float16*)(ws + offVh);

  cast_f32_to_f16<<<(DC * DN3 + 255) / 256, 256, 0, stream>>>(W_attn, Wah, (size_t)DC * DN3);
  cast_f32_to_f16<<<(DC * DC  + 255) / 256, 256, 0, stream>>>(W_proj, Wph, (size_t)DC * DC);

  dim3 gQkv(DN3 / 128, DM / 128);   // 24 x 32
  gemm_wmma<false><<<gQkv, 256, 0, stream>>>(x, Wah, b_attn, qkv, DM, DN3, DC);

  rope_split<<<(DB * DT * DH * (DHD / 2)) / 256, 256, 0, stream>>>(qkv, qhp, khp, vhp);

  dim3 gAtt(DT / 64, DB * DH);      // 32 x 32
  flash_attn<<<gAtt, 128, 0, stream>>>(qhp, khp, vhp, yh);

  dim3 gPrj(DC / 128, DM / 128);    // 8 x 32
  gemm_wmma<true><<<gPrj, 256, 0, stream>>>(yh, Wph, b_proj, out, DM, DC, DC);
}
